// TfLstmClassifier_48816598286560
// MI455X (gfx1250) — matrix operand order repaired and checked by reading, not compiled, not run on hardware
//
#include <hip/hip_runtime.h>

// ---------------- problem constants ----------------
#define BATCH 64
#define HID   512
#define SEQ   512
#define FOURH 2048
#define NCLS  10
#define HSZ   (BATCH * HID)   // 32768 elements per h buffer
#define NBLK  16
#define TPB   256

typedef __attribute__((ext_vector_type(16))) __bf16 v16bf;
typedef __attribute__((ext_vector_type(8)))  __bf16 v8bf;
typedef __attribute__((ext_vector_type(8)))  float  v8f;

// D = A(16x32 bf16) * B(32x16 bf16) + C(16x16 f32)
#define WMMA_BF16(a, b, c) \
  __builtin_amdgcn_wmma_f32_16x16x32_bf16(false, (a), false, (b), (short)0, (c), false, false)

__device__ __forceinline__ float fsig(float x)  { return 1.0f / (1.0f + __expf(-x)); }
__device__ __forceinline__ float ftanh(float x) { float e = __expf(2.0f * x); return (e - 1.0f) / (e + 1.0f); }

// Produce an opaque uniform 0 each call. Adding it to a weight pointer blocks
// LICM/CSE of the weight loads across the timestep loop (prevents the compiler
// hoisting 100s of weight VGPRs and spilling to scratch), while keeping the
// pointer chain rooted at the kernel argument so InferAddressSpaces still
// promotes the loads to global_load (not flat_load, which costs DScnt too).
__device__ __forceinline__ size_t opaque_zero() {
  size_t z = 0;
  asm volatile("" : "+s"(z));
  return z;
}

// A fragment: 16x32 bf16, M rows striped over lanes (contiguous k order, same as B; not the ISA table order):
// lane L: M = mbase + (L&15); holds K = kb..kb+15, kb = k0 + 16*(L>>4)
__device__ __forceinline__ v16bf load_a(const __bf16* __restrict__ h, int ldk,
                                        int mbase, int k0, int lane) {
  int M  = mbase + (lane & 15);
  int kb = k0 + ((lane >> 4) << 4);
  const __bf16* p = h + (size_t)M * ldk + kb;
  v8bf lo = *(const v8bf*)p;
  v8bf hi = *(const v8bf*)(p + 8);
  return __builtin_shufflevector(lo, hi, 0,1,2,3,4,5,6,7,8,9,10,11,12,13,14,15);
}

// ---------------- software grid barrier (16 blocks, counters in L2) ----------
__device__ __forceinline__ void grid_barrier(unsigned* cnt, unsigned* gen) {
  __threadfence();          // flush this thread's global stores to device scope
  __syncthreads();
  if (threadIdx.x == 0) {
    unsigned g = __atomic_load_n(gen, __ATOMIC_RELAXED);
    unsigned arrived = atomicAdd(cnt, 1u) + 1u;
    if (arrived == NBLK) {
      __atomic_store_n(cnt, 0u, __ATOMIC_RELAXED);
      __atomic_fetch_add(gen, 1u, __ATOMIC_RELEASE);
    } else {
      while (__atomic_load_n(gen, __ATOMIC_ACQUIRE) == g)
        __builtin_amdgcn_s_sleep(1);
    }
  }
  __syncthreads();
  __threadfence();          // acquire: see other blocks' stores
}

// ---------------- one-time prep: transpose + bf16-convert weights -----------
__global__ void prep_w0(const float* __restrict__ W0, __bf16* __restrict__ wt0) {
  size_t id = (size_t)blockIdx.x * blockDim.x + threadIdx.x;
  if (id >= (size_t)FOURH * HID) return;
  int n = (int)(id / HID), k = (int)(id % HID);
  wt0[id] = (__bf16)W0[(size_t)(1 + k) * FOURH + n];   // skip x row (row 0)
}
__global__ void prep_w1(const float* __restrict__ W1, __bf16* __restrict__ wt1) {
  size_t id = (size_t)blockIdx.x * blockDim.x + threadIdx.x;
  if (id >= (size_t)FOURH * 2 * HID) return;
  int n = (int)(id / (2 * HID)), k = (int)(id % (2 * HID));
  wt1[id] = (__bf16)W1[(size_t)k * FOURH + n];
}
__global__ void prep_misc(const float* __restrict__ W0, float* __restrict__ w0x,
                          __bf16* __restrict__ h1, __bf16* __restrict__ h2,
                          unsigned* __restrict__ bar) {
  int id = blockIdx.x * blockDim.x + threadIdx.x;
  if (id < HSZ) { h1[id] = (__bf16)0.0f; h2[id] = (__bf16)0.0f; }  // zero t=0 state (buffer 0)
  if (id < FOURH) w0x[id] = W0[id];                                // x-row of W0, kept f32
  if (id == 0) { bar[0] = 0u; bar[1] = 0u; }
}

// ---------------- persistent 2-layer LSTM + classifier ----------------------
__global__ __launch_bounds__(TPB, 2) void lstm_persistent(
    const float* __restrict__ x,  const float* __restrict__ b0,
    const float* __restrict__ b1, const float* __restrict__ sw,
    const float* __restrict__ sb,
    const __bf16* __restrict__ wt0, const __bf16* __restrict__ wt1,
    const float* __restrict__ w0x,
    __bf16* __restrict__ h1, __bf16* __restrict__ h2,
    unsigned* __restrict__ bar, float* __restrict__ out)
{
  const int lane  = threadIdx.x & 31;
  const int wv    = blockIdx.x * (TPB / 32) + (threadIdx.x >> 5); // 0..127
  const int m     = wv & 3;        // batch tile (16 rows)
  const int nh    = wv >> 2;       // h chunk (16 cols), 0..31
  const int khalf = lane >> 4;
  const int ncol  = nh * 16 + (lane & 15);   // 0..511
  const int mrow  = m * 16 + khalf * 8;      // base batch row of C/D rows this lane holds

  float c1[8], c2[8];
  #pragma unroll
  for (int r = 0; r < 8; ++r) { c1[r] = 0.0f; c2[r] = 0.0f; }

  const float bi1 = b0[ncol], bj1 = b0[HID + ncol], bf1 = b0[2 * HID + ncol], bo1 = b0[3 * HID + ncol];
  const float bi2 = b1[ncol], bj2 = b1[HID + ncol], bf2 = b1[2 * HID + ncol], bo2 = b1[3 * HID + ncol];
  const float wxi = w0x[ncol], wxj = w0x[HID + ncol], wxf = w0x[2 * HID + ncol], wxo = w0x[3 * HID + ncol];

  // per-lane element offsets into the transposed weight arrays
  const size_t boff1 = (size_t)ncol * HID + (khalf << 4);       // layer1: ldk = 512
  const size_t boff2 = (size_t)ncol * (2 * HID) + (khalf << 4); // layer2: ldk = 1024

  for (int t = 0; t < SEQ; ++t) {
    const __bf16* h1p = h1 + (size_t)(t & 1) * HSZ;
    __bf16*       h1n = h1 + (size_t)((t + 1) & 1) * HSZ;
    const __bf16* h2p = h2 + (size_t)(t & 1) * HSZ;
    __bf16*       h2n = h2 + (size_t)((t + 1) & 1) * HSZ;

    // opaque per-timestep offset: blocks LICM of weight loads across the t
    // loop, but keeps the pointers rooted at kernel args (=> global_load)
    const size_t z  = opaque_zero();
    const __bf16* bb1 = wt0 + z + boff1;
    const __bf16* bb2 = wt1 + z + boff2;

    // ---- layer 1: z = h1_prev @ W0h  (K = 512), all 4 gates for this tile
    v8f zi = {}, zj = {}, zf = {}, zo = {};
    for (int k0 = 0; k0 < HID; k0 += 32) {
      v16bf a = load_a(h1p, HID, m * 16, k0, lane);
      zi = WMMA_BF16(a, *(const v16bf*)(bb1 + k0 + 0 * (size_t)HID * HID), zi);
      zj = WMMA_BF16(a, *(const v16bf*)(bb1 + k0 + 1 * (size_t)HID * HID), zj);
      zf = WMMA_BF16(a, *(const v16bf*)(bb1 + k0 + 2 * (size_t)HID * HID), zf);
      zo = WMMA_BF16(a, *(const v16bf*)(bb1 + k0 + 3 * (size_t)HID * HID), zo);
    }
    #pragma unroll
    for (int r = 0; r < 8; ++r) {
      int b = mrow + r;
      float xv = x[(size_t)b * SEQ + t];                  // scalar input term (rank-1)
      float ig = fsig (zi[r] + bi1 + xv * wxi);
      float jg = ftanh(zj[r] + bj1 + xv * wxj);
      float fg = fsig (zf[r] + bf1 + xv * wxf + 1.0f);    // forget bias
      float og = fsig (zo[r] + bo1 + xv * wxo);
      c1[r] = c1[r] * fg + ig * jg;                       // cell state stays in VGPRs
      h1n[(size_t)b * HID + ncol] = (__bf16)(ftanh(c1[r]) * og);
    }
    grid_barrier(bar, bar + 1);   // h1_t visible to all 16 WGPs

    // ---- layer 2: z = [h1_t ; h2_prev] @ W1  (K = 1024)
    v8f yi = {}, yj = {}, yf = {}, yo = {};
    for (int k0 = 0; k0 < 2 * HID; k0 += 32) {
      const __bf16* asrc = (k0 < HID) ? h1n : h2p;
      int ak = (k0 < HID) ? k0 : (k0 - HID);
      v16bf a = load_a(asrc, HID, m * 16, ak, lane);
      yi = WMMA_BF16(a, *(const v16bf*)(bb2 + k0 + 0 * (size_t)HID * 2 * HID), yi);
      yj = WMMA_BF16(a, *(const v16bf*)(bb2 + k0 + 1 * (size_t)HID * 2 * HID), yj);
      yf = WMMA_BF16(a, *(const v16bf*)(bb2 + k0 + 2 * (size_t)HID * 2 * HID), yf);
      yo = WMMA_BF16(a, *(const v16bf*)(bb2 + k0 + 3 * (size_t)HID * 2 * HID), yo);
    }
    #pragma unroll
    for (int r = 0; r < 8; ++r) {
      int b = mrow + r;
      float ig = fsig (yi[r] + bi2);
      float jg = ftanh(yj[r] + bj2);
      float fg = fsig (yf[r] + bf2 + 1.0f);
      float og = fsig (yo[r] + bo2);
      c2[r] = c2[r] * fg + ig * jg;
      h2n[(size_t)b * HID + ncol] = (__bf16)(ftanh(c2[r]) * og);
    }
    grid_barrier(bar, bar + 1);   // h2_t visible before next step
  }

  // ---- classifier head: logits = h2_T @ softmax_w + softmax_b  (tiny, block 0)
  if (blockIdx.x == 0) {
    const __bf16* hf = h2;   // SEQ even -> final h2 landed in buffer 0
    for (int idx = threadIdx.x; idx < BATCH * NCLS; idx += TPB) {
      int b = idx / NCLS, c = idx % NCLS;
      float acc = sb[c];
      for (int k = 0; k < HID; ++k)
        acc += (float)hf[(size_t)b * HID + k] * sw[(size_t)k * NCLS + c];
      out[idx] = acc;
    }
  }
}

// ---------------- host launcher ---------------------------------------------
extern "C" void kernel_launch(void* const* d_in, const int* in_sizes, int n_in,
                              void* d_out, int out_size, void* d_ws, size_t ws_size,
                              hipStream_t stream) {
  (void)in_sizes; (void)n_in; (void)out_size; (void)ws_size;
  const float* x  = (const float*)d_in[0];   // [64, 512]
  const float* W0 = (const float*)d_in[1];   // [513, 2048]
  const float* b0 = (const float*)d_in[2];   // [2048]
  const float* W1 = (const float*)d_in[3];   // [1024, 2048]
  const float* b1 = (const float*)d_in[4];   // [2048]
  const float* sw = (const float*)d_in[5];   // [512, 10]
  const float* sb = (const float*)d_in[6];   // [10]
  float* out = (float*)d_out;                // [64, 10]

  char* ws = (char*)d_ws;
  __bf16*   wt0 = (__bf16*)(ws);                                   // 2 MiB  [2048][512]
  __bf16*   wt1 = (__bf16*)(ws + (2ull << 20));                    // 4 MiB  [2048][1024]
  float*    w0x = (float*)  (ws + (6ull << 20));                   // 8 KiB
  __bf16*   h1  = (__bf16*)(ws + (6ull << 20) + (16ull << 10));    // 128 KiB (2 bufs)
  __bf16*   h2  = (__bf16*)(ws + (6ull << 20) + (144ull << 10));   // 128 KiB (2 bufs)
  unsigned* bar = (unsigned*)(ws + (6ull << 20) + (272ull << 10)); // counters

  hipLaunchKernelGGL(prep_w0, dim3((FOURH * HID + TPB - 1) / TPB), dim3(TPB), 0, stream, W0, wt0);
  hipLaunchKernelGGL(prep_w1, dim3((FOURH * 2 * HID + TPB - 1) / TPB), dim3(TPB), 0, stream, W1, wt1);
  hipLaunchKernelGGL(prep_misc, dim3((HSZ + TPB - 1) / TPB), dim3(TPB), 0, stream,
                     W0, w0x, h1, h2, bar);
  hipLaunchKernelGGL(lstm_persistent, dim3(NBLK), dim3(TPB), 0, stream,
                     x, b0, b1, sw, sb, wt0, wt1, w0x, h1, h2, bar, out);
}
